// LMUCell_87282325390061
// MI455X (gfx1250) — matrix operand order repaired and checked by reading, not compiled, not run on hardware
//
#include <hip/hip_runtime.h>
#include <math.h>

// ---------------- problem constants ----------------
#define BB     64
#define TT     512
#define DIN    256
#define UNITS  1024
#define ORDER  256
#define KCAT   1536        // DIN + UNITS + ORDER (GEMM reduction length)
#define AROW   2816        // DIN + 2*UNITS + 2*ORDER (double-buffered h and m)
#define NWG    32
#define TPB    256

typedef __attribute__((ext_vector_type(16))) __bf16 v16bf;
typedef __attribute__((ext_vector_type(8)))  float  v8f;

// round-to-nearest-even f32 -> bf16
__device__ __forceinline__ unsigned short f2bf(float f) {
  unsigned int u = __float_as_uint(f);
  u += 0x7FFFu + ((u >> 16) & 1u);
  return (unsigned short)(u >> 16);
}

// A-fragment (16x32 bf16): lane holds row M=lane&15; lanes<16 carry K {0..15},
// lanes>=16 carry K {16..31}. p0 already includes the +16-halfword lane select.
__device__ __forceinline__ v16bf load_a_frag(const unsigned short* p0) {
  union { unsigned int w[8]; v16bf v; } fr;
  const unsigned int* a = (const unsigned int*)(p0);        // 8 halfs @ K base
  const unsigned int* b = (const unsigned int*)(p0 + 8);    // 8 halfs @ K base+8
#pragma unroll
  for (int i = 0; i < 4; ++i) { fr.w[i] = a[i]; fr.w[4 + i] = b[i]; }
  return fr.v;
}

// grid-wide barrier over NWG persistent blocks (atomic counter, monotone phase)
__device__ __forceinline__ void grid_sync(int* syncc, int tgt) {
  __threadfence();
  __syncthreads();
  if (threadIdx.x == 0) {
    atomicAdd(syncc, 1);
    while (__hip_atomic_load(syncc, __ATOMIC_ACQUIRE, __HIP_MEMORY_SCOPE_AGENT) < tgt)
      __builtin_amdgcn_s_sleep(1);
  }
  __syncthreads();
  __threadfence();
}

// ---------------- init kernel: weight packing + state zeroing ----------------
// Wg layout: [wg(32)][nt(2)][n(16)][k(1536)] bf16, column n_global = wg*32+nt*16+n
// ATg layout: [wgm(16)][n(16)][k(256)] bf16, column c = wgm*16+n  (AT transposed)
__global__ void k_init(const float* __restrict__ ik, const float* __restrict__ hk,
                       const float* __restrict__ mk, const float* __restrict__ AT,
                       unsigned short* Wg, unsigned short* ATg, unsigned short* Act,
                       float* Mbuf, float* Upart, int* syncc) {
  const long stride = (long)gridDim.x * blockDim.x;
  long gid = (long)blockIdx.x * blockDim.x + threadIdx.x;

  for (long i = gid; i < (long)NWG * 2 * 16 * KCAT; i += stride) {
    int k   = (int)(i % KCAT);
    int r   = (int)(i / KCAT);
    int n   = r % 16;  r /= 16;
    int nt  = r % 2;   r /= 2;
    int wg  = r;
    int col = wg * 32 + nt * 16 + n;
    float v;
    if (k < DIN)              v = ik[(size_t)k * UNITS + col];
    else if (k < DIN + UNITS) v = hk[(size_t)(k - DIN) * UNITS + col];
    else                      v = mk[(size_t)(k - DIN - UNITS) * UNITS + col];
    Wg[i] = f2bf(v);
  }
  for (long i = gid; i < 16L * 16 * ORDER; i += stride) {
    int k   = (int)(i % ORDER);
    int n   = (int)((i / ORDER) % 16);
    int wgm = (int)(i / (ORDER * 16));
    ATg[i] = f2bf(AT[(size_t)k * ORDER + wgm * 16 + n]);
  }
  for (long i = gid; i < (long)BB * AROW; i += stride) Act[i] = 0;
  for (long i = gid; i < 2L * BB * ORDER; i += stride) Mbuf[i] = 0.f;
  for (long i = gid; i < (long)NWG * BB; i += stride) Upart[i] = 0.f;
  if (gid == 0) syncc[0] = 0;
}

// ---------------- persistent recurrent kernel ----------------
__launch_bounds__(TPB)
__global__ void k_lmu(const float* __restrict__ inputs, const float* __restrict__ ie,
                      const float* __restrict__ he, const float* __restrict__ me,
                      const float* __restrict__ BT,
                      const unsigned short* __restrict__ Wg,
                      const unsigned short* __restrict__ ATg,
                      unsigned short* Act, float* Mbuf, float* Upart,
                      float* out, int* syncc) {
  __shared__ __align__(32) unsigned short Wlds[2 * 16 * KCAT];   // 96 KB column slice
  __shared__ __align__(32) unsigned short ATlds[16 * ORDER];     //  8 KB AT columns
  __shared__ float red[TPB];
  __shared__ float u_lds[BB];
  __shared__ float hred[BB];

  const int tid  = threadIdx.x;
  const int wg   = blockIdx.x;
  const int lane = tid & 31;
  const int wv   = tid >> 5;
  const int am   = lane & 15;
  const int asel = (lane & 16) ? 16 : 0;   // A-fragment K select
  const int bsel = (lane & 16) ? 16 : 0;   // B-fragment K select
  const int mo   = (lane & 16) ? 8  : 0;   // C-fragment row select

  // Pin this WG's weight slice in LDS for the whole sequence (HBM read once).
  for (int i = tid; i < 2 * 16 * KCAT; i += TPB)
    Wlds[i] = Wg[(size_t)wg * (2 * 16 * KCAT) + i];
  if (wg < 16)
    for (int i = tid; i < 16 * ORDER; i += TPB)
      ATlds[i] = ATg[(size_t)wg * (16 * ORDER) + i];
  __syncthreads();

  int ph = 0;
  for (int t = 0; t < TT; ++t) {
    const int wsel = t & 1;        // state buffer written at step t
    const int rsel = wsel ^ 1;     // state buffer holding step t-1

    // ======== barrier: h_{t-1}, Upart, Mbuf[rsel], Act[rsel] all published ========
    ph++; grid_sync(syncc, ph * NWG);

    // ======== Phase AB: stage x_t ; rebuild u ; m_t = m + m@AT + u*BT ========
    {
      // stage x_t (2 batch rows per WG, bf16 into Act)
      int rowsel = tid >> 7;
      int bx = 2 * wg + rowsel;
      int l = tid & 127;
      const float* xr = inputs + ((size_t)bx * TT + t) * DIN;
#pragma unroll
      for (int d = l; d < DIN; d += 128) Act[(size_t)bx * AROW + d] = f2bf(xr[d]);
      if (t + 1 < TT) __builtin_prefetch(xr + DIN, 0, 1);   // global_prefetch next x_t
    }
    if (wg < 16) {
      // u_b = x.ie + m.me + sum_w Upart[w][b]   (f32 sources; redundant over 16 WGs)
      int r = tid >> 2, q = tid & 3;
      float acc = 0.f;
      const float* xrow = inputs + ((size_t)r * TT + t) * DIN;
      for (int d = q; d < DIN; d += 4) acc += xrow[d] * ie[d];
      const float* mrow = Mbuf + ((size_t)rsel * BB + r) * ORDER;
      for (int c = q; c < ORDER; c += 4) acc += mrow[c] * me[c];
      for (int w = q; w < NWG; w += 4) acc += Upart[(size_t)w * BB + r];
      red[tid] = acc;
      __syncthreads();
      if (q == 0) u_lds[r] = red[tid] + red[tid + 1] + red[tid + 2] + red[tid + 3];
      __syncthreads();

      // m-update: 16 columns per WG, waves 0..3 take one 16-row tile each
      if (wv < 4) {
        const int mt = wv;
        const unsigned short* arow = Act + (size_t)(mt * 16 + am) * AROW
                                         + (DIN + 2 * UNITS + rsel * ORDER) + asel;
        const unsigned short* bbase = &ATlds[(size_t)am * ORDER];
        v8f macc = {};
#pragma unroll
        for (int kb = 0; kb < ORDER; kb += 32) {
          v16bf a  = load_a_frag(arow + kb);
          v16bf bf = *(const v16bf*)(bbase + kb + bsel);
          macc = __builtin_amdgcn_wmma_f32_16x16x32_bf16(false, a, false, bf,
                                                         (short)0, macc, false, false);
        }
        const int c = wg * 16 + am;
        const float btc = BT[c];
#pragma unroll
        for (int i = 0; i < 8; ++i) {
          int b = mt * 16 + mo + i;
          float mnew = macc[i] + Mbuf[((size_t)rsel * BB + b) * ORDER + c]
                               + u_lds[b] * btc;
          Mbuf[((size_t)wsel * BB + b) * ORDER + c] = mnew;          // f32 master state
          Act[(size_t)b * AROW + (DIN + 2 * UNITS + wsel * ORDER) + c] = f2bf(mnew);
        }
      }
    }

    // ======== barrier: x_t staged, m_t published ========
    ph++; grid_sync(syncc, ph * NWG);

    // ======== Phase C: h_t = tanh([x_t | h_{t-1} | m_t] @ Wcat) ========
    {
      const int mt = wv & 3;
      const int nt = wv >> 2;
      const unsigned short* arow  = Act + (size_t)(mt * 16 + am) * AROW + asel;
      const unsigned short* bbase = &Wlds[(size_t)(nt * 16 + am) * KCAT];
      v8f acc = {};
      for (int kb = 0; kb < KCAT; kb += 32) {
        int col = (kb < DIN) ? kb
                : (kb < DIN + UNITS) ? (kb + rsel * UNITS)      // h_{t-1} buffer
                : (kb + UNITS + wsel * ORDER);                  // m_t buffer
        v16bf a  = load_a_frag(arow + col);
        v16bf bf = *(const v16bf*)(bbase + kb + bsel);
        acc = __builtin_amdgcn_wmma_f32_16x16x32_bf16(false, a, false, bf,
                                                      (short)0, acc, false, false);
      }
      if (tid < BB) hred[tid] = 0.f;
      __syncthreads();
      const int ucol = wg * 32 + nt * 16 + am;
      const float hew = he[ucol];
#pragma unroll
      for (int i = 0; i < 8; ++i) {
        int b = mt * 16 + mo + i;
        float hval = tanhf(acc[i]);
        out[((size_t)b * TT + t) * UNITS + ucol] = hval;             // f32 h master
        Act[(size_t)b * AROW + DIN + wsel * UNITS + ucol] = f2bf(hval);
        atomicAdd(&hred[b], hval * hew);                             // ds_add_f32
      }
      __syncthreads();
      // per-WG partial of h_t . hidden_encoders for next step's u
      if (tid < BB) Upart[(size_t)wg * BB + tid] = hred[tid];
    }
  }
}

// ---------------- host launch ----------------
extern "C" void kernel_launch(void* const* d_in, const int* in_sizes, int n_in,
                              void* d_out, int out_size, void* d_ws, size_t ws_size,
                              hipStream_t stream) {
  const float* inputs = (const float*)d_in[0];
  const float* ie     = (const float*)d_in[1];
  const float* he     = (const float*)d_in[2];
  const float* me     = (const float*)d_in[3];
  const float* ik     = (const float*)d_in[4];
  const float* hk     = (const float*)d_in[5];
  const float* mk     = (const float*)d_in[6];
  const float* AT     = (const float*)d_in[7];
  const float* BT     = (const float*)d_in[8];
  float* out = (float*)d_out;
  char*  ws  = (char*)d_ws;

  const size_t offWg  = 0;
  const size_t offATg = offWg  + (size_t)NWG * 2 * 16 * KCAT * sizeof(unsigned short); // 3,145,728
  const size_t offAct = offATg + (size_t)16 * 16 * ORDER * sizeof(unsigned short);     // +131,072
  const size_t offM   = offAct + (size_t)BB * AROW * sizeof(unsigned short);           // +360,448
  const size_t offUp  = offM   + (size_t)2 * BB * ORDER * sizeof(float);               // +131,072
  const size_t offS   = offUp  + (size_t)NWG * BB * sizeof(float);                     // +8,192

  unsigned short* Wg    = (unsigned short*)(ws + offWg);
  unsigned short* ATg   = (unsigned short*)(ws + offATg);
  unsigned short* Act   = (unsigned short*)(ws + offAct);
  float*          Mbuf  = (float*)(ws + offM);
  float*          Upart = (float*)(ws + offUp);
  int*            sy    = (int*)(ws + offS);

  k_init<<<3072, TPB, 0, stream>>>(ik, hk, mk, AT, Wg, ATg, Act, Mbuf, Upart, sy);
  k_lmu<<<NWG, TPB, 0, stream>>>(inputs, ie, he, me, BT, Wg, ATg, Act, Mbuf, Upart, out, sy);
}
